// GraphStructuredSparseAttention_34626026340707
// MI455X (gfx1250) — matrix operand order repaired and checked by reading, not compiled, not run on hardware
//
#include <hip/hip_runtime.h>
#include <math.h>

typedef unsigned short u16;
typedef __attribute__((ext_vector_type(16))) __bf16 v16bf;
typedef __attribute__((ext_vector_type(8)))  float  v8f;

union FragBF { uint4 u[2]; v16bf v; };

__device__ __forceinline__ u16 f32_to_bf16(float f) {
  unsigned int u = __float_as_uint(f);
  u += 0x7fffu + ((u >> 16) & 1u);   // round-to-nearest-even
  return (u16)(u >> 16);
}

// ---------------- Tensor Data Mover support --------------------------------
#if defined(__HIP_DEVICE_COMPILE__) && __has_builtin(__builtin_amdgcn_tensor_load_to_lds)
#define USE_TDM 1
#else
#define USE_TDM 0
#endif

#if USE_TDM
typedef __attribute__((ext_vector_type(4))) unsigned int u32x4;
typedef __attribute__((ext_vector_type(8))) int i32x8;
typedef __attribute__((ext_vector_type(4))) int i32x4;

// 2-D tile load: rows x 32 bf16 elements, row-major source with stride_elems,
// LDS destination padded 16B after every 64B row (=> 80B pitch = 40 u16).
__device__ __forceinline__ void tdm_load_tile(unsigned lds_byte_addr,
                                              const void* gptr,
                                              unsigned rows,
                                              unsigned stride_elems) {
  const unsigned long long ga = (unsigned long long)gptr;
  u32x4 g0;
  g0[0] = 1u;                                             // count=1 (valid D#)
  g0[1] = lds_byte_addr;                                  // lds_addr
  g0[2] = (unsigned)ga;                                   // global_addr[31:0]
  g0[3] = ((unsigned)(ga >> 32) & 0x01ffffffu) | (2u << 30); // addr[56:32] | type=2
  i32x8 g1;
  // data_size=2B | pad_enable | pad_interval=16 DW | pad_amount=4 DW
  g1[0] = (int)((1u << 16) | (1u << 20) | (3u << 22) | (3u << 25));
  g1[1] = 0;                                  // abar addr=0, tensor_dim0[15:0]=0
  g1[2] = 0x4000;                             // tensor_dim0 = 0x40000000 (no OOB)
  g1[3] = (int)(0x4000u | (32u << 16));       // tensor_dim1 = 0x40000000, tile_dim0=32
  g1[4] = (int)rows;                          // tile_dim1 = rows, tile_dim2=0
  g1[5] = (int)stride_elems;                  // tensor_dim0_stride[31:0]
  g1[6] = 0;                                  // stride hi, dim1_stride lo
  g1[7] = 0;
  const i32x4 z4 = {0, 0, 0, 0};
#if defined(__clang_major__) && (__clang_major__ >= 23)
  const i32x8 z8 = {0, 0, 0, 0, 0, 0, 0, 0};
  __builtin_amdgcn_tensor_load_to_lds(g0, g1, z4, z4, z8, 0);
#else
  __builtin_amdgcn_tensor_load_to_lds(g0, g1, z4, z4, 0);
#endif
}
#endif  // USE_TDM

// ---------------- fp32 -> bf16 (x4 vectorized, grid-stride) ----------------
__global__ void cvt_f32_bf16_kernel(const float* __restrict__ in,
                                    u16* __restrict__ out, size_t n4) {
  size_t i = (size_t)blockIdx.x * blockDim.x + threadIdx.x;
  const size_t stride = (size_t)gridDim.x * blockDim.x;
  const float4* in4 = (const float4*)in;
  uint2* out4 = (uint2*)out;
  for (; i < n4; i += stride) {
    float4 a = in4[i];
    uint2 r;
    r.x = (unsigned)f32_to_bf16(a.x) | ((unsigned)f32_to_bf16(a.y) << 16);
    r.y = (unsigned)f32_to_bf16(a.z) | ((unsigned)f32_to_bf16(a.w) << 16);
    out4[i] = r;
  }
}

// ---------------- bf16 NT-GEMM: C[m,n] = sum_k A[m,k] * Bsrc[n,k] ----------
// MODE 0: bf16 out (+bias[n] if non-null), row-major
// MODE 1: bf16 out (+bias[n]), transposed store C[n*ldc + m]  (for V^T)
// MODE 2: fp32 out = acc*scale + mask*(-1e9), row-major        (scores)
// MODE 3: fp32 out = acc + bias[n], row-major                  (final)
#define BM 128
#define BN 64
#define BK 32
#define LDS_STRIDE 40  // 32 + 8 pad (u16) -> 80B pitch; matches TDM pad descriptor

template <int MODE>
__global__ __launch_bounds__(256)
void gemm_bf16_kernel(const u16* __restrict__ A, long long aBatch, int lda,
                      const u16* __restrict__ Bsrc, long long bBatch, int ldb,
                      void* __restrict__ Cout, long long cBatch, int ldc,
                      int K,
                      const float* __restrict__ bias,
                      const int* __restrict__ mask, long long maskBatch,
                      float scale) {
  __shared__ u16 sA[2][BM * LDS_STRIDE];
  __shared__ u16 sB[2][BN * LDS_STRIDE];

  const int tid   = threadIdx.x;
  const int lane  = tid & 31;
  const int wid   = tid >> 5;
  const int l16   = lane & 15;
  const int h     = lane >> 4;     // half-wave id
  const int waveM = wid & 3;       // 4 waves along M (32 rows each)
  const int waveN = wid >> 2;      // 2 waves along N (32 cols each)

  const int blockM = blockIdx.y * BM;
  const int blockN = blockIdx.x * BN;
  const int b      = blockIdx.z;

  const u16* Ab = A + (size_t)b * aBatch + (size_t)blockM * lda;
  const u16* Bb = Bsrc + (size_t)b * bBatch + (size_t)blockN * ldb;

  v8f acc[2][2] = {};
  const int nk = K / BK;

#if USE_TDM
  // ---- async tensor staging: wave 0 drives the Tensor Data Mover ----------
  const unsigned ldsA[2] = {(unsigned)(size_t)&sA[0][0], (unsigned)(size_t)&sA[1][0]};
  const unsigned ldsB[2] = {(unsigned)(size_t)&sB[0][0], (unsigned)(size_t)&sB[1][0]};
  if (wid == 0) {  // prologue: tile 0 -> buffer 0
    tdm_load_tile(ldsA[0], Ab, BM, (unsigned)lda);
    tdm_load_tile(ldsB[0], Bb, BN, (unsigned)ldb);
  }
#else
  const int arow = tid >> 2;        // 0..63
  const int aseg = (tid & 3) * 8;   // k-element offset of this thread's 16B chunk
  uint4 ra0, ra1, rb;
  {  // prologue: k-tile 0 -> LDS buffer 0
    const u16* pa = Ab + aseg;
    ra0 = *(const uint4*)(pa + (size_t)arow * lda);
    ra1 = *(const uint4*)(pa + (size_t)(arow + 64) * lda);
    rb  = *(const uint4*)(Bb + (size_t)arow * ldb + aseg);
  }
  *(uint4*)&sA[0][arow * LDS_STRIDE + aseg]        = ra0;
  *(uint4*)&sA[0][(arow + 64) * LDS_STRIDE + aseg] = ra1;
  *(uint4*)&sB[0][arow * LDS_STRIDE + aseg]        = rb;
#endif

  for (int t = 0; t < nk; ++t) {
    const int cur = t & 1;
#if USE_TDM
    __syncthreads();  // everyone done reading buf (cur^1) from iteration t-1
    if (wid == 0) {
      if (t + 1 < nk) {
        const int k0 = (t + 1) * BK;
        tdm_load_tile(ldsA[cur ^ 1], Ab + k0, BM, (unsigned)lda);
        tdm_load_tile(ldsB[cur ^ 1], Bb + k0, BN, (unsigned)ldb);
        __builtin_amdgcn_s_wait_tensorcnt(2);  // tile t done (TDM is in-order)
      } else {
        __builtin_amdgcn_s_wait_tensorcnt(0);
      }
    }
    __syncthreads();  // tile t visible to all waves
#else
    __syncthreads();
    if (t + 1 < nk) {  // prefetch next k-tile into registers
      const int k0 = (t + 1) * BK;
      const u16* pa = Ab + k0 + aseg;
      ra0 = *(const uint4*)(pa + (size_t)arow * lda);
      ra1 = *(const uint4*)(pa + (size_t)(arow + 64) * lda);
      rb  = *(const uint4*)(Bb + (size_t)arow * ldb + k0 + aseg);
    }
#endif

    // A fragment (16-bit 16x32 layout): lane half h reads K chunks [8h..8h+7]
    // and [16+8h..16+8h+7] of row (l16); B fragment: column-lane reads the
    // same K chunks [8h..8h+7] and [16+8h..16+8h+7] of Bsrc row (l16).
    FragBF fa[2], fb[2];
#pragma unroll
    for (int i = 0; i < 2; ++i) {
      const int row = waveM * 32 + i * 16 + l16;
      const u16* p = &sA[cur][row * LDS_STRIDE];
      fa[i].u[0] = *(const uint4*)(p + 8 * h);
      fa[i].u[1] = *(const uint4*)(p + 16 + 8 * h);
    }
#pragma unroll
    for (int j = 0; j < 2; ++j) {
      const int col = waveN * 32 + j * 16 + l16;
      const u16* p = &sB[cur][col * LDS_STRIDE];
      fb[j].u[0] = *(const uint4*)(p + 8 * h);
      fb[j].u[1] = *(const uint4*)(p + 16 + 8 * h);
    }

#pragma unroll
    for (int i = 0; i < 2; ++i)
#pragma unroll
      for (int j = 0; j < 2; ++j)
        acc[i][j] = __builtin_amdgcn_wmma_f32_16x16x32_bf16(
            false, fa[i].v, false, fb[j].v, (short)0, acc[i][j], false, false);

#if !USE_TDM
    if (t + 1 < nk) {  // stage prefetched tile into the other buffer
      const int nxt = cur ^ 1;
      *(uint4*)&sA[nxt][arow * LDS_STRIDE + aseg]        = ra0;
      *(uint4*)&sA[nxt][(arow + 64) * LDS_STRIDE + aseg] = ra1;
      *(uint4*)&sB[nxt][arow * LDS_STRIDE + aseg]        = rb;
    }
#endif
  }

  // ------------- epilogue (C/D layout: VGPR r -> M = i*16 + 8h + r) --------
#pragma unroll
  for (int i = 0; i < 2; ++i)
#pragma unroll
    for (int j = 0; j < 2; ++j) {
      const int n = blockN + waveN * 32 + j * 16 + l16;
#pragma unroll
      for (int r = 0; r < 8; ++r) {
        const int m = blockM + waveM * 32 + i * 16 + h * 8 + r;
        float v = acc[i][j][r];
        if constexpr (MODE == 0) {
          if (bias) v += bias[n];
          ((u16*)Cout)[(size_t)b * cBatch + (size_t)m * ldc + n] = f32_to_bf16(v);
        } else if constexpr (MODE == 1) {
          if (bias) v += bias[n];
          ((u16*)Cout)[(size_t)b * cBatch + (size_t)n * ldc + m] = f32_to_bf16(v);
        } else if constexpr (MODE == 2) {
          const int mk = mask[(size_t)b * maskBatch + (size_t)m * ldc + n];
          ((float*)Cout)[(size_t)b * cBatch + (size_t)m * ldc + n] =
              v * scale + (float)mk * -1e9f;
        } else {
          if (bias) v += bias[n];
          ((float*)Cout)[(size_t)b * cBatch + (size_t)m * ldc + n] = v;
        }
      }
    }
}

// ---------------- per-row sigmoid gate + softmax ---------------------------
__device__ __forceinline__ float block_reduce(float val, float* buf, bool is_max) {
  const int t = threadIdx.x;
  buf[t] = val;
  __syncthreads();
  for (int s = 128; s > 0; s >>= 1) {
    if (t < s) buf[t] = is_max ? fmaxf(buf[t], buf[t + s]) : (buf[t] + buf[t + s]);
    __syncthreads();
  }
  float r = buf[0];
  __syncthreads();
  return r;
}

__global__ __launch_bounds__(256)
void row_gate_softmax_kernel(const float* __restrict__ S,
                             const float* __restrict__ We,
                             const float* __restrict__ be,
                             u16* __restrict__ P, int N) {
  __shared__ float red[256];
  const int row = blockIdx.x, b = blockIdx.y;
  const size_t base = ((size_t)b * N + row) * (size_t)N;
  const float* s = S + base;

  float v[8];
  float dot = 0.f, mx = -3.0e38f;
#pragma unroll
  for (int i = 0; i < 8; ++i) {
    const int k = threadIdx.x + i * 256;
    const float x = s[k];
    v[i] = x;
    dot += x * We[k];
    mx = fmaxf(mx, x);
  }
  dot = block_reduce(dot, red, false);
  mx  = block_reduce(mx, red, true);

  const float g = 1.f / (1.f + __expf(-(dot + be[0])));  // sigmoid gate > 0
  const float m = g * mx;                                 // max(g*s) = g*max(s)

  float e[8], sum = 0.f;
#pragma unroll
  for (int i = 0; i < 8; ++i) { e[i] = __expf(g * v[i] - m); sum += e[i]; }
  sum = block_reduce(sum, red, false);
  const float inv = 1.f / sum;
#pragma unroll
  for (int i = 0; i < 8; ++i)
    P[base + threadIdx.x + i * 256] = f32_to_bf16(e[i] * inv);
}

// ---------------- launch ---------------------------------------------------
extern "C" void kernel_launch(void* const* d_in, const int* in_sizes, int n_in,
                              void* d_out, int out_size, void* d_ws, size_t ws_size,
                              hipStream_t stream) {
  (void)in_sizes; (void)n_in; (void)out_size; (void)ws_size;
  const int E = 2048, N = 2048, B = 4;

  const float* q   = (const float*)d_in[0];
  const float* k   = (const float*)d_in[1];
  const float* v   = (const float*)d_in[2];
  const int*   msk = (const int*)d_in[3];
  const float* Wq  = (const float*)d_in[4];
  const float* bq  = (const float*)d_in[5];
  const float* Wk  = (const float*)d_in[6];
  const float* bk  = (const float*)d_in[7];
  const float* Wv  = (const float*)d_in[8];
  const float* bv  = (const float*)d_in[9];
  const float* We  = (const float*)d_in[10];
  const float* be  = (const float*)d_in[11];
  const float* Wo  = (const float*)d_in[12];
  const float* bo  = (const float*)d_in[13];
  float* out = (float*)d_out;

  const size_t NE  = (size_t)N * E;   // per-batch matrix elems
  const size_t BNE = (size_t)B * NE;
  const size_t EE  = (size_t)E * E;
  const long long NEll = (long long)NE;
  const long long NN   = (long long)N * N;

  // workspace layout (u16 elems), with safe lifetime-based aliasing
  u16* xq  = (u16*)d_ws;
  u16* xk  = xq + BNE;
  u16* xv  = xk + BNE;
  u16* wqb = xv + BNE;
  u16* wkb = wqb + EE;
  u16* wvb = wkb + EE;
  u16* wob = wvb + EE;
  u16* Qb  = wob + EE;
  u16* Kb  = Qb + BNE;
  u16* Vt  = Kb + BNE;
  float* S = (float*)xq;  // B*N*N fp32 == bytes of xq+xk (dead after K proj)
  u16* Pb  = xv;          // dead after V proj
  u16* Ob  = Qb;          // dead after scores

  const dim3 blk(256);
  cvt_f32_bf16_kernel<<<4096, blk, 0, stream>>>(q,  xq,  BNE / 4);
  cvt_f32_bf16_kernel<<<4096, blk, 0, stream>>>(k,  xk,  BNE / 4);
  cvt_f32_bf16_kernel<<<4096, blk, 0, stream>>>(v,  xv,  BNE / 4);
  cvt_f32_bf16_kernel<<<1024, blk, 0, stream>>>(Wq, wqb, EE / 4);
  cvt_f32_bf16_kernel<<<1024, blk, 0, stream>>>(Wk, wkb, EE / 4);
  cvt_f32_bf16_kernel<<<1024, blk, 0, stream>>>(Wv, wvb, EE / 4);
  cvt_f32_bf16_kernel<<<1024, blk, 0, stream>>>(Wo, wob, EE / 4);

  const float scale = 1.0f / sqrtf((float)E);
  const dim3 gProj(E / BN, N / BM, B);   // 32 x 16 x 4
  const dim3 gScore(N / BN, N / BM, B);

  // Q = xq @ Wq^T + bq ; K likewise ; V stored transposed (Vt[b][e][n])
  gemm_bf16_kernel<0><<<gProj, blk, 0, stream>>>(xq, NEll, E, wqb, 0, E, Qb, NEll, E, E, bq, nullptr, 0, 1.f);
  gemm_bf16_kernel<0><<<gProj, blk, 0, stream>>>(xk, NEll, E, wkb, 0, E, Kb, NEll, E, E, bk, nullptr, 0, 1.f);
  gemm_bf16_kernel<1><<<gProj, blk, 0, stream>>>(xv, NEll, E, wvb, 0, E, Vt, NEll, N, E, bv, nullptr, 0, 1.f);

  // S = Q K^T * scale + mask * -1e9
  gemm_bf16_kernel<2><<<gScore, blk, 0, stream>>>(Qb, NEll, E, Kb, NEll, E, S, NN, N, E, nullptr, msk, NN, scale);

  // edge gate + softmax -> P (bf16)
  row_gate_softmax_kernel<<<dim3(N, B), blk, 0, stream>>>(S, We, be, Pb, N);

  // O = P @ V   (Bsrc = Vt, rows are V columns)
  gemm_bf16_kernel<0><<<gProj, blk, 0, stream>>>(Pb, NN, N, Vt, NEll, N, Ob, NEll, E, N, nullptr, nullptr, 0, 1.f);

  // out = O @ Wo^T + bo  (fp32)
  gemm_bf16_kernel<3><<<gProj, blk, 0, stream>>>(Ob, NEll, E, wob, 0, E, out, NEll, E, E, bo, nullptr, 0, 1.f);
}
